// TransformerBlock_60756607369302
// MI455X (gfx1250) — matrix operand order repaired and checked by reading, not compiled, not run on hardware
//
#include <hip/hip_runtime.h>
#include <math.h>

#define BCONST 4
#define SSEQ   2048
#define DMODEL 768
#define NHEAD  12
#define DHEAD  64
#define FDIM   3072
#define NTOK   (BCONST * SSEQ)

typedef __bf16 bf16_t;
typedef __attribute__((ext_vector_type(16))) __bf16 v16bf;
typedef __attribute__((ext_vector_type(8)))  __bf16 bf16x8;
typedef __attribute__((ext_vector_type(8)))  float  v8f;
typedef __attribute__((ext_vector_type(4)))  unsigned int v4u;
typedef __attribute__((ext_vector_type(8)))  unsigned int v8u;

union Frag16 { v16bf v; bf16x8 h[2]; };

static __device__ inline bf16_t f2bf(float f) { return (bf16_t)f; }

static __device__ inline v8f zero8() {
    v8f z;
    #pragma unroll
    for (int i = 0; i < 8; i++) z[i] = 0.0f;
    return z;
}

// ---------------------------------------------------------------------------
// Tensor Data Mover: 2-D tile load Global -> LDS (bf16 elements).
// D# group0/group1 per CDNA5 ISA 8.3/8.4; 2-group form (VADDR2/3 = NULL).
// Pad: after every 16 dwords (one 32-elem tile row) insert 4 dwords -> LDS
// row stride of 40 bf16, matching the fragment-load layout below.
// Issued by one wave; caller must s_wait_tensorcnt + barrier.
// ---------------------------------------------------------------------------
static __device__ inline void tdm_load_2d_pad(const void* gaddr, unsigned lds_addr,
                                              unsigned tensor_d0, unsigned tensor_d1,
                                              unsigned tile_d0, unsigned tile_d1,
                                              unsigned stride_elems)
{
    const unsigned long long ga = (unsigned long long)(uintptr_t)gaddr;
    v4u g0;
    g0[0] = 1u;                                   // count=1, user descriptor
    g0[1] = lds_addr;                             // LDS byte address
    g0[2] = (unsigned)ga;                         // global_addr[31:0]
    g0[3] = (unsigned)(ga >> 32) | (2u << 30);    // global_addr[56:32] | type=2
    v8u g1;
    g1[0] = (1u << 16)                            // data_size = 2 bytes
          | (1u << 20)                            // pad_enable
          | (3u << 22)                            // pad_interval = 16 dwords
          | (3u << 25);                           // pad_amount   = 4 dwords
    g1[1] = (tensor_d0 & 0xFFFFu) << 16;          // tensor_dim0[15:0] @ bits63:48
    g1[2] = (tensor_d0 >> 16) | ((tensor_d1 & 0xFFFFu) << 16);
    g1[3] = (tensor_d1 >> 16) | (tile_d0 << 16);  // tile_dim0 @ bits127:112
    g1[4] = tile_d1;                              // tile_dim1 (tile_dim2 = 0)
    g1[5] = stride_elems;                         // tensor_dim0_stride[31:0]
    g1[6] = 0u;
    g1[7] = 0u;
    asm volatile("tensor_load_to_lds %0, %1" :: "s"(g0), "s"(g1) : "memory");
}

// ---------------------------------------------------------------------------
// Generic bf16 WMMA GEMM: C[M,N] = A[M,K] * B[K,N] + bias, tiled 128x128x32.
// 256 threads = 8 wave32; wave grid 2 (M) x 4 (N); each wave -> 64x32 C tile.
// A tile staged by TDM (tensor_load_to_lds, wave 0); B tile staged transposed
// by all threads. Double-buffered LDS, one barrier per K-step.
// epi: 0 = fp32 out, 2 = exact-GELU -> bf16 out, 3 = QKV scatter (q,k,v^T bf16)
// ---------------------------------------------------------------------------
__global__ __launch_bounds__(256)
void gemm_bf16_wmma(const bf16_t* __restrict__ A, const bf16_t* __restrict__ Bm,
                    const float* __restrict__ bias, int M, int N, int K, int epi,
                    float* __restrict__ outF, bf16_t* __restrict__ outB,
                    bf16_t* __restrict__ qo, bf16_t* __restrict__ ko,
                    bf16_t* __restrict__ vto)
{
    __shared__ alignas(16) bf16_t As[2][128][40];   // [buf][row][k] (TDM-padded)
    __shared__ alignas(16) bf16_t Bs[2][128][40];   // [buf][col][k] (transposed)

    const int tid  = threadIdx.x;
    const int lane = tid & 31;
    const int wave = tid >> 5;
    const int wm   = wave & 1;        // 0..1 : 64-row slab
    const int wn   = wave >> 1;       // 0..3 : 32-col slab
    const int bm0  = blockIdx.y * 128;
    const int bn0  = blockIdx.x * 128;
    const int lh   = lane & 15;
    const bool hi  = lane >= 16;

    const unsigned ldsA[2] = { (unsigned)(uintptr_t)(void*)&As[0][0][0],
                               (unsigned)(uintptr_t)(void*)&As[1][0][0] };

    v8f acc[4][2];
    #pragma unroll
    for (int i = 0; i < 4; i++)
        #pragma unroll
        for (int j = 0; j < 2; j++) acc[i][j] = zero8();

    const int brow = tid >> 3;          // 0..31  (k within tile)
    const int bseg = (tid & 7) * 16;    // 0..112 (col segment)

    const int NK = K / 32;

    // ---- prologue: stage k-tile 0 into buffer 0 ----
    if (wave == 0)
        tdm_load_2d_pad(A + (size_t)bm0 * K, ldsA[0],
                        (unsigned)K, (unsigned)(M - bm0), 32u, 128u, (unsigned)K);
    {
        const bf16_t* gp = Bm + (size_t)brow * N + bn0 + bseg;
        bf16x8 v0 = *(const bf16x8*)(gp);
        bf16x8 v1 = *(const bf16x8*)(gp + 8);
        #pragma unroll
        for (int i = 0; i < 8; i++) Bs[0][bseg + i][brow]     = v0[i];
        #pragma unroll
        for (int i = 0; i < 8; i++) Bs[0][bseg + 8 + i][brow] = v1[i];
    }
    if (wave == 0) __builtin_amdgcn_s_wait_tensorcnt(0);
    __syncthreads();

    for (int kt = 0; kt < NK; ++kt) {
        const int cur = kt & 1;

        // ---- stage next k-tile into the other buffer (overlapped) ----
        if (kt + 1 < NK) {
            const int kn = (kt + 1) * 32;
            if (wave == 0)
                tdm_load_2d_pad(A + (size_t)bm0 * K + kn, ldsA[cur ^ 1],
                                (unsigned)(K - kn), (unsigned)(M - bm0),
                                32u, 128u, (unsigned)K);
            const bf16_t* gp = Bm + (size_t)(kn + brow) * N + bn0 + bseg;
            bf16x8 v0 = *(const bf16x8*)(gp);
            bf16x8 v1 = *(const bf16x8*)(gp + 8);
            #pragma unroll
            for (int i = 0; i < 8; i++) Bs[cur ^ 1][bseg + i][brow]     = v0[i];
            #pragma unroll
            for (int i = 0; i < 8; i++) Bs[cur ^ 1][bseg + 8 + i][brow] = v1[i];
            if (kt + 2 < NK)
                __builtin_prefetch(Bm + (size_t)((kt + 2) * 32 + brow) * N + bn0 + bseg,
                                   0, 0);
        }

        // ---- compute from current buffer ----
        // B fragments (lane n = column; K contiguous, kb = hi?16:0)
        Frag16 bfr[2];
        #pragma unroll
        for (int nt = 0; nt < 2; nt++) {
            const bf16_t* bp = &Bs[cur][wn * 32 + nt * 16 + lh][hi ? 16 : 0];
            bfr[nt].h[0] = *(const bf16x8*)(bp);
            bfr[nt].h[1] = *(const bf16x8*)(bp + 8);
        }
        // A fragments (lane = row M; K contiguous kb..kb+15, kb = hi?16:0)
        #pragma unroll
        for (int mt = 0; mt < 4; mt++) {
            Frag16 afr;
            const bf16_t* ap = &As[cur][wm * 64 + mt * 16 + lh][hi ? 16 : 0];
            afr.h[0] = *(const bf16x8*)(ap);
            afr.h[1] = *(const bf16x8*)(ap + 8);
            #pragma unroll
            for (int nt = 0; nt < 2; nt++)
                acc[mt][nt] = __builtin_amdgcn_wmma_f32_16x16x32_bf16(
                    false, afr.v, false, bfr[nt].v, (short)0, acc[mt][nt],
                    false, false);
        }

        if (wave == 0 && kt + 1 < NK) __builtin_amdgcn_s_wait_tensorcnt(0);
        __syncthreads();
    }

    // --- epilogue ---
    #pragma unroll
    for (int mt = 0; mt < 4; mt++) {
        #pragma unroll
        for (int nt = 0; nt < 2; nt++) {
            #pragma unroll
            for (int r = 0; r < 8; r++) {
                const int row = bm0 + wm * 64 + mt * 16 + r + (hi ? 8 : 0);
                const int col = bn0 + wn * 32 + nt * 16 + lh;
                float v = acc[mt][nt][r] + bias[col];
                if (epi == 0) {
                    outF[(size_t)row * N + col] = v;
                } else if (epi == 2) {
                    float g = 0.5f * v * (1.0f + erff(v * 0.70710678118654752f));
                    outB[(size_t)row * N + col] = f2bf(g);
                } else { // epi == 3 : QKV scatter
                    const int b = row / SSEQ, s = row % SSEQ;
                    if (col < DMODEL) {
                        const int hh = col >> 6, e = col & 63;
                        qo[(((size_t)b * NHEAD + hh) * SSEQ + s) * DHEAD + e] = f2bf(v);
                    } else if (col < 2 * DMODEL) {
                        const int c = col - DMODEL, hh = c >> 6, e = c & 63;
                        ko[(((size_t)b * NHEAD + hh) * SSEQ + s) * DHEAD + e] = f2bf(v);
                    } else { // v stored transposed [B,H,Dh,S]
                        const int c = col - 2 * DMODEL, hh = c >> 6, e = c & 63;
                        vto[(((size_t)b * NHEAD + hh) * DHEAD + e) * SSEQ + s] = f2bf(v);
                    }
                }
            }
        }
    }
}

// ---------------------------------------------------------------------------
// Flash attention (causal): 4 waves/block, 64 query rows, 32-kv tiles.
// q,k: bf16 [B,H,S,64]; vT: bf16 [B,H,64,S]; ctx out: bf16 [NTOK, D].
// ---------------------------------------------------------------------------
__global__ __launch_bounds__(128)
void flash_attn(const bf16_t* __restrict__ q, const bf16_t* __restrict__ kk,
                const bf16_t* __restrict__ vt, bf16_t* __restrict__ ctx)
{
    __shared__ alignas(16) bf16_t P[4][16][32];   // per-wave P staging (C->A layout)

    const int lane  = threadIdx.x & 31;
    const int wave  = threadIdx.x >> 5;
    const int lh    = lane & 15;
    const bool hiL  = lane >= 16;
    const int qbase = blockIdx.x * 64;
    const int hh    = blockIdx.y;
    const int b     = blockIdx.z;

    const bf16_t* qh = q  + ((size_t)(b * NHEAD + hh)) * SSEQ * DHEAD;
    const bf16_t* kh = kk + ((size_t)(b * NHEAD + hh)) * SSEQ * DHEAD;
    const bf16_t* vh = vt + ((size_t)(b * NHEAD + hh)) * DHEAD * SSEQ;

    const int qr  = qbase + wave * 16 + lh;       // A-frag row for this lane
    const int kbA = hiL ? 16 : 0;
    Frag16 qa0, qa1;                               // Q rows, K = dims 0..31 / 32..63
    qa0.h[0] = *(const bf16x8*)(qh + (size_t)qr * DHEAD + kbA);
    qa0.h[1] = *(const bf16x8*)(qh + (size_t)qr * DHEAD + kbA + 8);
    qa1.h[0] = *(const bf16x8*)(qh + (size_t)qr * DHEAD + 32 + kbA);
    qa1.h[1] = *(const bf16x8*)(qh + (size_t)qr * DHEAD + 32 + kbA + 8);

    v8f o[4];
    #pragma unroll
    for (int i = 0; i < 4; i++) o[i] = zero8();
    float m[8], l[8];
    #pragma unroll
    for (int r = 0; r < 8; r++) { m[r] = -1e30f; l[r] = 0.0f; }

    const int Ml     = hiL ? 8 : 0;
    const int kbB    = hiL ? 16 : 0;
    const int nTiles = qbase / 32 + 2;             // causal upper bound for block

    for (int j = 0; j < nTiles; j++) {
        const int kvb = j * 32;
        // scores S = Q * K^T : 16x32 per wave, accumulated over Dh=64
        v8f c[2];
        #pragma unroll
        for (int nt = 0; nt < 2; nt++) {
            const int kv = kvb + nt * 16 + lh;     // B-frag column = kv row of K
            const bf16_t* kp = kh + (size_t)kv * DHEAD;
            Frag16 kf0, kf1;
            kf0.h[0] = *(const bf16x8*)(kp + kbB);
            kf0.h[1] = *(const bf16x8*)(kp + kbB + 8);
            kf1.h[0] = *(const bf16x8*)(kp + 32 + kbB);
            kf1.h[1] = *(const bf16x8*)(kp + 32 + kbB + 8);
            v8f ct = zero8();
            ct = __builtin_amdgcn_wmma_f32_16x16x32_bf16(false, qa0.v, false, kf0.v,
                                                         (short)0, ct, false, false);
            ct = __builtin_amdgcn_wmma_f32_16x16x32_bf16(false, qa1.v, false, kf1.v,
                                                         (short)0, ct, false, false);
            c[nt] = ct;
        }
        // online softmax (per C-fragment row r; half-wave = consistent row set)
        #pragma unroll
        for (int r = 0; r < 8; r++) {
            const int qrow = qbase + wave * 16 + r + Ml;
            const int kv0 = kvb + lh, kv1 = kvb + 16 + lh;
            float s0 = (kv0 <= qrow) ? c[0][r] * 0.125f : -1e30f;
            float s1 = (kv1 <= qrow) ? c[1][r] * 0.125f : -1e30f;
            float t = fmaxf(s0, s1);
            t = fmaxf(t, __shfl_xor(t, 1));
            t = fmaxf(t, __shfl_xor(t, 2));
            t = fmaxf(t, __shfl_xor(t, 4));
            t = fmaxf(t, __shfl_xor(t, 8));
            const float mn = fmaxf(m[r], t);
            const float al = __expf(m[r] - mn);
            const float p0 = __expf(s0 - mn);
            const float p1 = __expf(s1 - mn);
            float rs = p0 + p1;
            rs += __shfl_xor(rs, 1);
            rs += __shfl_xor(rs, 2);
            rs += __shfl_xor(rs, 4);
            rs += __shfl_xor(rs, 8);
            l[r] = l[r] * al + rs;
            m[r] = mn;
            #pragma unroll
            for (int t4 = 0; t4 < 4; t4++) o[t4][r] *= al;
            P[wave][r + Ml][lh]      = f2bf(p0);
            P[wave][r + Ml][lh + 16] = f2bf(p1);
        }
        __syncthreads();
        // P (A layout) and V^T (B layout), ctx += P * V
        Frag16 pa;
        pa.h[0] = *(const bf16x8*)&P[wave][lh][kbA];
        pa.h[1] = *(const bf16x8*)&P[wave][lh][kbA + 8];
        #pragma unroll
        for (int t4 = 0; t4 < 4; t4++) {
            const bf16_t* vp = vh + (size_t)(t4 * 16 + lh) * SSEQ + kvb + kbB;
            Frag16 vf;
            vf.h[0] = *(const bf16x8*)(vp);
            vf.h[1] = *(const bf16x8*)(vp + 8);
            o[t4] = __builtin_amdgcn_wmma_f32_16x16x32_bf16(false, pa.v, false, vf.v,
                                                            (short)0, o[t4], false, false);
        }
        __syncthreads();
    }

    // write ctx (concat heads) as bf16
    #pragma unroll
    for (int t4 = 0; t4 < 4; t4++) {
        #pragma unroll
        for (int r = 0; r < 8; r++) {
            const int row   = qbase + wave * 16 + r + Ml;
            const int token = b * SSEQ + row;
            const int col   = hh * DHEAD + t4 * 16 + lh;
            ctx[(size_t)token * DMODEL + col] = f2bf(o[t4][r] / l[r]);
        }
    }
}

// ---------------------------------------------------------------------------
// Fused residual + LayerNorm: out = LN(a+b)*w + bias ; optional bf16 mirror.
// One block per token row (D = 768 = 3 * 256).
// ---------------------------------------------------------------------------
__global__ __launch_bounds__(256)
void add_ln(const float* __restrict__ a, const float* __restrict__ bsrc,
            const float* __restrict__ w, const float* __restrict__ bb,
            float* __restrict__ outF, bf16_t* __restrict__ outB)
{
    __shared__ float red[256];
    const int row = blockIdx.x;
    const float* ar = a    + (size_t)row * DMODEL;
    const float* br = bsrc + (size_t)row * DMODEL;
    float v[3];
    float s = 0.0f;
    #pragma unroll
    for (int i = 0; i < 3; i++) {
        const int c = threadIdx.x + i * 256;
        v[i] = ar[c] + br[c];
        s += v[i];
    }
    red[threadIdx.x] = s;
    __syncthreads();
    for (int off = 128; off; off >>= 1) {
        if (threadIdx.x < off) red[threadIdx.x] += red[threadIdx.x + off];
        __syncthreads();
    }
    const float mean = red[0] * (1.0f / DMODEL);
    __syncthreads();
    float s2 = 0.0f;
    #pragma unroll
    for (int i = 0; i < 3; i++) { const float d = v[i] - mean; s2 += d * d; }
    red[threadIdx.x] = s2;
    __syncthreads();
    for (int off = 128; off; off >>= 1) {
        if (threadIdx.x < off) red[threadIdx.x] += red[threadIdx.x + off];
        __syncthreads();
    }
    const float rstd = rsqrtf(red[0] * (1.0f / DMODEL) + 1e-5f);
    #pragma unroll
    for (int i = 0; i < 3; i++) {
        const int c = threadIdx.x + i * 256;
        const float y = (v[i] - mean) * rstd * w[c] + bb[c];
        if (outF) outF[(size_t)row * DMODEL + c] = y;
        if (outB) outB[(size_t)row * DMODEL + c] = f2bf(y);
    }
}

// ---------------------------------------------------------------------------
// casts / packing
// ---------------------------------------------------------------------------
__global__ void cast_bf16(const float* __restrict__ src, bf16_t* __restrict__ dst, int n)
{
    const int i = blockIdx.x * 256 + threadIdx.x;
    if (i < n) dst[i] = f2bf(src[i]);
}

// Wq/Wk/Wv are [H, D, Dh]; pack to row-major [D, 3D] with col = h*Dh + e.
__global__ void pack_wqkv(const float* __restrict__ Wq, const float* __restrict__ Wk,
                          const float* __restrict__ Wv, bf16_t* __restrict__ dst)
{
    const int i = blockIdx.x * 256 + threadIdx.x;
    if (i >= DMODEL * 3 * DMODEL) return;
    const int d = i / (3 * DMODEL);
    const int c = i % (3 * DMODEL);
    const float* W;
    int cc;
    if (c < DMODEL)          { W = Wq; cc = c; }
    else if (c < 2 * DMODEL) { W = Wk; cc = c - DMODEL; }
    else                     { W = Wv; cc = c - 2 * DMODEL; }
    const int hh = cc >> 6, e = cc & 63;
    dst[i] = f2bf(W[((size_t)hh * DMODEL + d) * DHEAD + e]);
}

__global__ void pack_bias(const float* __restrict__ bq, const float* __restrict__ bk,
                          const float* __restrict__ bv, float* __restrict__ dst)
{
    const int i = blockIdx.x * 256 + threadIdx.x;
    if (i >= 3 * DMODEL) return;
    dst[i] = (i < DMODEL) ? bq[i]
           : (i < 2 * DMODEL) ? bk[i - DMODEL] : bv[i - 2 * DMODEL];
}

// ---------------------------------------------------------------------------
extern "C" void kernel_launch(void* const* d_in, const int* in_sizes, int n_in,
                              void* d_out, int out_size, void* d_ws, size_t ws_size,
                              hipStream_t stream)
{
    const float* x    = (const float*)d_in[0];
    const float* Wq   = (const float*)d_in[1];
    const float* bq   = (const float*)d_in[2];
    const float* Wk   = (const float*)d_in[3];
    const float* bk   = (const float*)d_in[4];
    const float* Wv   = (const float*)d_in[5];
    const float* bv   = (const float*)d_in[6];
    const float* Wo   = (const float*)d_in[7];
    const float* bo   = (const float*)d_in[8];
    const float* W1   = (const float*)d_in[9];
    const float* b1   = (const float*)d_in[10];
    const float* W2   = (const float*)d_in[11];
    const float* b2   = (const float*)d_in[12];
    const float* ln1w = (const float*)d_in[13];
    const float* ln1b = (const float*)d_in[14];
    const float* ln2w = (const float*)d_in[15];
    const float* ln2b = (const float*)d_in[16];

    char* p = (char*)d_ws;
    auto alloc = [&](size_t bytes) -> char* {
        char* r = p;
        p += (bytes + 255) & ~(size_t)255;
        return r;
    };
    bf16_t* xb   = (bf16_t*)alloc((size_t)NTOK * DMODEL * 2);
    bf16_t* wqkv = (bf16_t*)alloc((size_t)DMODEL * 3 * DMODEL * 2);
    float*  bqkv = (float*) alloc((size_t)3 * DMODEL * 4);
    bf16_t* wob  = (bf16_t*)alloc((size_t)DMODEL * DMODEL * 2);
    bf16_t* w1b  = (bf16_t*)alloc((size_t)DMODEL * FDIM * 2);
    bf16_t* w2b  = (bf16_t*)alloc((size_t)FDIM * DMODEL * 2);
    bf16_t* qbuf = (bf16_t*)alloc((size_t)NTOK * DMODEL * 2);
    bf16_t* kbuf = (bf16_t*)alloc((size_t)NTOK * DMODEL * 2);
    bf16_t* vtb  = (bf16_t*)alloc((size_t)NTOK * DMODEL * 2);
    bf16_t* ctx  = (bf16_t*)alloc((size_t)NTOK * DMODEL * 2);
    float*  tmp  = (float*) alloc((size_t)NTOK * DMODEL * 4);
    float*  hbuf = (float*) alloc((size_t)NTOK * DMODEL * 4);
    bf16_t* hb   = (bf16_t*)alloc((size_t)NTOK * DMODEL * 2);
    bf16_t* gbuf = (bf16_t*)alloc((size_t)NTOK * FDIM * 2);

    cast_bf16<<<(NTOK * DMODEL + 255) / 256, 256, 0, stream>>>(x, xb, NTOK * DMODEL);
    pack_wqkv<<<(DMODEL * 3 * DMODEL + 255) / 256, 256, 0, stream>>>(Wq, Wk, Wv, wqkv);
    pack_bias<<<(3 * DMODEL + 255) / 256, 256, 0, stream>>>(bq, bk, bv, bqkv);
    cast_bf16<<<(DMODEL * DMODEL + 255) / 256, 256, 0, stream>>>(Wo, wob, DMODEL * DMODEL);
    cast_bf16<<<(DMODEL * FDIM + 255) / 256, 256, 0, stream>>>(W1, w1b, DMODEL * FDIM);
    cast_bf16<<<(FDIM * DMODEL + 255) / 256, 256, 0, stream>>>(W2, w2b, FDIM * DMODEL);

    // QKV projection: [8192,768] x [768,2304] -> q,k,v^T scatter (bf16)
    gemm_bf16_wmma<<<dim3(3 * DMODEL / 128, NTOK / 128), 256, 0, stream>>>(
        xb, wqkv, bqkv, NTOK, 3 * DMODEL, DMODEL, 3,
        nullptr, nullptr, qbuf, kbuf, vtb);

    // causal flash attention -> ctx (bf16, heads concatenated)
    flash_attn<<<dim3(SSEQ / 64, NHEAD, BCONST), dim3(128), 0, stream>>>(
        qbuf, kbuf, vtb, ctx);

    // output projection: ctx @ Wo + bo -> fp32 tmp
    gemm_bf16_wmma<<<dim3(DMODEL / 128, NTOK / 128), 256, 0, stream>>>(
        ctx, wob, bo, NTOK, DMODEL, DMODEL, 0,
        tmp, nullptr, nullptr, nullptr, nullptr);

    // h = LN(x + att_out); keep fp32 + bf16 copies
    add_ln<<<NTOK, 256, 0, stream>>>(x, tmp, ln1w, ln1b, hbuf, hb);

    // FFN up + exact GELU -> bf16
    gemm_bf16_wmma<<<dim3(FDIM / 128, NTOK / 128), 256, 0, stream>>>(
        hb, w1b, b1, NTOK, FDIM, DMODEL, 2,
        nullptr, gbuf, nullptr, nullptr, nullptr);

    // FFN down -> fp32 tmp
    gemm_bf16_wmma<<<dim3(DMODEL / 128, NTOK / 128), 256, 0, stream>>>(
        gbuf, w2b, b2, NTOK, DMODEL, FDIM, 0,
        tmp, nullptr, nullptr, nullptr, nullptr);

    // out = LN(h + ff)
    add_ln<<<NTOK, 256, 0, stream>>>(hbuf, tmp, ln2w, ln2b, (float*)d_out, nullptr);
}
